// MHSA_3049426780457
// MI455X (gfx1250) — matrix operand order repaired and checked by reading, not compiled, not run on hardware
//
#include <hip/hip_runtime.h>

// ---------------------------------------------------------------------------
// MHSA with relative-position bias, MI455X (gfx1250, wave32, WMMA).
// Pipeline: xT cast -> relT build -> QKV projection (WMMA f16, 16x32 tile) ->
//           flash-style attention (WMMA f16, f32 accum, online softmax,
//           32 query rows per wave to double WMMA per B-operand load).
// ---------------------------------------------------------------------------

typedef __attribute__((ext_vector_type(16))) _Float16 v16h;
typedef __attribute__((ext_vector_type(8)))  _Float16 v8h;
typedef __attribute__((ext_vector_type(8)))  float    v8f;
typedef __attribute__((ext_vector_type(4)))  float    v4f;

static constexpr int Bc    = 16;    // batch
static constexpr int Cc    = 256;   // channels
static constexpr int HEADS = 4;
static constexpr int Dd    = 64;    // head dim
static constexpr int Nn    = 1024;  // W*H

// --- helpers: build v16h WMMA operands from two contiguous 16B runs --------
__device__ __forceinline__ v16h cat16(v8h lo, v8h hi) {
  v16h r;
#pragma unroll
  for (int e = 0; e < 8; ++e) { r[e] = lo[e]; r[8 + e] = hi[e]; }
  return r;
}
// A operand (16x32 f16): lane holds K runs [lo..lo+7] and [lo+16..lo+23]
__device__ __forceinline__ v16h loadA16(const _Float16* p) {
  return cat16(*(const v8h*)p, *(const v8h*)(p + 16));
}
// B operand (32x16 f16): lane holds K runs [lo..lo+7] and [lo+16..lo+23]
__device__ __forceinline__ v16h loadB16(const _Float16* p) {
  return cat16(*(const v8h*)p, *(const v8h*)(p + 16));
}

// ---------------------------------------------------------------------------
// Kernel 1: xT[b][n][c] (f16) <- x[b][c][n] (f32).  Coalesced reads.
// ---------------------------------------------------------------------------
__global__ void k_cast_xT(const float* __restrict__ x, _Float16* __restrict__ xT) {
  size_t f = (size_t)blockIdx.x * blockDim.x + threadIdx.x;  // == x flat idx
  int n = (int)(f % Nn);
  int c = (int)((f / Nn) % Cc);
  int b = (int)(f / ((size_t)Nn * Cc));
  xT[((size_t)b * Nn + n) * Cc + c] = (_Float16)x[f];
}

// ---------------------------------------------------------------------------
// Kernel 2: relT[h][n][d] (f16) = rel_h[h][d][ih] + rel_w[h][d][iw], n=iw*32+ih
// ---------------------------------------------------------------------------
__global__ void k_build_rel(const float* __restrict__ rel_h,
                            const float* __restrict__ rel_w,
                            _Float16* __restrict__ relT) {
  int f = blockIdx.x * blockDim.x + threadIdx.x;  // == relT flat idx (h,n,d)
  int dd = f % Dd;
  int n  = (f / Dd) % Nn;
  int h  = f / (Dd * Nn);
  int ih = n & 31, iw = n >> 5;
  relT[f] = (_Float16)(rel_h[(h * Dd + dd) * 32 + ih] +
                       rel_w[(h * Dd + dd) * 32 + iw]);
}

// ---------------------------------------------------------------------------
// Kernel 3: QKV projection.  One wave computes a 16(o) x 32(n) tile so each
// converted weight A-tile feeds two WMMAs.  K=256 in 8 steps.
// grid = (B*N/32, C/16, 3), block = 32.
// Writes: which=0 -> QT[b][h][n][d], 1 -> KT[b][h][n][d], 2 -> Vd[b][h][d][n]
// ---------------------------------------------------------------------------
__global__ void k_proj(const float* __restrict__ Wq, const float* __restrict__ bq,
                       const float* __restrict__ Wk, const float* __restrict__ bk,
                       const float* __restrict__ Wv, const float* __restrict__ bv,
                       const _Float16* __restrict__ xT,
                       _Float16* __restrict__ QT, _Float16* __restrict__ KT,
                       _Float16* __restrict__ Vd) {
  const int lane = threadIdx.x;
  const int half = lane >> 4, l16 = lane & 15;
  const int which = blockIdx.z;
  const float* Wm = (which == 0) ? Wq : (which == 1) ? Wk : Wv;
  const float* bm = (which == 0) ? bq : (which == 1) ? bk : bv;

  const int pt = blockIdx.x;          // position tile over (b, n)
  const int b  = pt >> 5;             // 32 n-tiles (of 32) per batch
  const int n0 = (pt & 31) << 5;
  const int o0 = blockIdx.y << 4;

  v8f acc[2] = {{}, {}};
  const float*    wrow  = Wm + (size_t)(o0 + l16) * Cc;
  const _Float16* xrow0 = xT + ((size_t)b * Nn + (n0 + l16)) * Cc;
  const _Float16* xrow1 = xrow0 + (size_t)16 * Cc;

#pragma unroll
  for (int kc = 0; kc < Cc; kc += 32) {
    // A: W rows (f32 -> f16), runs at kc+half*8 and +16
    const float* wp = wrow + kc + half * 8;
    v4f w0 = *(const v4f*)(wp);      v4f w1 = *(const v4f*)(wp + 4);
    v4f w2 = *(const v4f*)(wp + 16); v4f w3 = *(const v4f*)(wp + 20);
    v16h a;
#pragma unroll
    for (int e = 0; e < 4; ++e) {
      a[e]      = (_Float16)w0[e];  a[4 + e]  = (_Float16)w1[e];
      a[8 + e]  = (_Float16)w2[e];  a[12 + e] = (_Float16)w3[e];
    }
    // B: xT rows (K runs at kc+half*8 and +16), two n-subtiles share A
    v16h b0 = loadB16(xrow0 + kc + half * 8);
    v16h b1 = loadB16(xrow1 + kc + half * 8);
    acc[0] = __builtin_amdgcn_wmma_f32_16x16x32_f16(false, a, false, b0,
                                                    (short)0, acc[0], false, false);
    acc[1] = __builtin_amdgcn_wmma_f32_16x16x32_f16(false, a, false, b1,
                                                    (short)0, acc[1], false, false);
  }

#pragma unroll
  for (int nt = 0; nt < 2; ++nt) {
#pragma unroll
    for (int v = 0; v < 8; ++v) {
      const int o = o0 + v + half * 8;           // D-tile row
      const int n = n0 + nt * 16 + l16;          // D-tile col
      const float y = acc[nt][v] + bm[o];
      const int h = o >> 6, dd = o & 63;
      if (which == 0)
        QT[(((size_t)b * HEADS + h) * Nn + n) * Dd + dd] = (_Float16)y;
      else if (which == 1)
        KT[(((size_t)b * HEADS + h) * Nn + n) * Dd + dd] = (_Float16)y;
      else
        Vd[(((size_t)b * HEADS + h) * Dd + dd) * Nn + n] = (_Float16)y;
    }
  }
}

// ---------------------------------------------------------------------------
// Kernel 4: flash attention.  One wave owns 32 query rows (two 16-row
// i-subtiles) of one (b,h), so every B-operand tile feeds two WMMAs.
// Loop j in steps of 32: logits = Q_i . K_j + rel_i . Q_j, online softmax
// (m,l per row), O += P @ V^T.  24 WMMA per step, 12 B-tile loads.
// grid = B*HEADS*(N/32), block = 32.
// ---------------------------------------------------------------------------
__global__ void k_attn(const _Float16* __restrict__ QT,
                       const _Float16* __restrict__ KT,
                       const _Float16* __restrict__ Vd,
                       const _Float16* __restrict__ relT,
                       float* __restrict__ out) {
  __shared__ __align__(16) _Float16 sP[2][16 * 32];  // attn tile staging

  const int lane = threadIdx.x;
  const int half = lane >> 4, l16 = lane & 15;
  const int wg = blockIdx.x;
  const int i0 = (wg & 31) << 5;        // 32 row-tiles of 32 per (b,h)
  const int h  = (wg >> 5) & (HEADS - 1);
  const int b  = wg >> 7;

  const _Float16* qbase = QT + ((size_t)b * HEADS + h) * Nn * Dd;
  const _Float16* kbase = KT + ((size_t)b * HEADS + h) * Nn * Dd;
  const _Float16* vbase = Vd + ((size_t)b * HEADS + h) * Dd * Nn;
  const _Float16* rbase = relT + (size_t)h * Nn * Dd;

  // A operands held for the whole j loop: Q^T rows i, rel^T rows i (K=d)
  v16h Qa[2][2], Ra[2][2];
#pragma unroll
  for (int it = 0; it < 2; ++it) {
    const size_t row = (size_t)(i0 + it * 16 + l16) * Dd;
#pragma unroll
    for (int t = 0; t < 2; ++t) {
      Qa[it][t] = loadA16(qbase + row + t * 32 + half * 8);
      Ra[it][t] = loadA16(rbase + row + t * 32 + half * 8);
    }
  }

  float m[2][8], l[2][8];
  v8f Oacc[2][4] = {{{}, {}, {}, {}}, {{}, {}, {}, {}}};
#pragma unroll
  for (int it = 0; it < 2; ++it)
#pragma unroll
    for (int v = 0; v < 8; ++v) { m[it][v] = -1e30f; l[it][v] = 0.0f; }

  for (int j0 = 0; j0 < Nn; j0 += 32) {
    // ---- logits: S[it][jc] covers rows i0+it*16.., cols j0+jc*16.. -------
    v8f S[2][2] = {{{}, {}}, {{}, {}}};
#pragma unroll
    for (int jc = 0; jc < 2; ++jc) {
      const int j = j0 + jc * 16 + l16;  // this lane's column (position)
#pragma unroll
      for (int t = 0; t < 2; ++t) {
        const v16h Bk = loadB16(kbase + (size_t)j * Dd + t * 32 + half * 8);
        const v16h Bq = loadB16(qbase + (size_t)j * Dd + t * 32 + half * 8);
#pragma unroll
        for (int it = 0; it < 2; ++it) {
          S[it][jc] = __builtin_amdgcn_wmma_f32_16x16x32_f16(
              false, Qa[it][t], false, Bk, (short)0, S[it][jc], false, false);
          S[it][jc] = __builtin_amdgcn_wmma_f32_16x16x32_f16(
              false, Ra[it][t], false, Bq, (short)0, S[it][jc], false, false);
        }
      }
    }

    // ---- online softmax: row r = v + half*8 lives in 16 lanes of a half --
#pragma unroll
    for (int it = 0; it < 2; ++it) {
#pragma unroll
      for (int v = 0; v < 8; ++v) {
        float rmax = fmaxf(S[it][0][v], S[it][1][v]);
        rmax = fmaxf(rmax, __shfl_xor(rmax, 1));
        rmax = fmaxf(rmax, __shfl_xor(rmax, 2));
        rmax = fmaxf(rmax, __shfl_xor(rmax, 4));
        rmax = fmaxf(rmax, __shfl_xor(rmax, 8));
        const float mn = fmaxf(m[it][v], rmax);
        const float sc = __expf(m[it][v] - mn);
        const float p0 = __expf(S[it][0][v] - mn);
        const float p1 = __expf(S[it][1][v] - mn);
        float rs = p0 + p1;
        rs += __shfl_xor(rs, 1);
        rs += __shfl_xor(rs, 2);
        rs += __shfl_xor(rs, 4);
        rs += __shfl_xor(rs, 8);
        l[it][v] = l[it][v] * sc + rs;
        m[it][v] = mn;
        S[it][0][v] = p0; S[it][1][v] = p1;   // reuse S as P
#pragma unroll
        for (int t = 0; t < 4; ++t) Oacc[it][t][v] *= sc;
      }
    }

    // ---- permute P (D-tile layout) -> A-operand layout via LDS ----------
#pragma unroll
    for (int it = 0; it < 2; ++it) {
#pragma unroll
      for (int v = 0; v < 8; ++v) {
        const int r = v + half * 8;
        sP[it][r * 32 + l16]      = (_Float16)S[it][0][v];
        sP[it][r * 32 + 16 + l16] = (_Float16)S[it][1][v];
      }
    }
    __syncthreads();
    v16h Pa[2];
#pragma unroll
    for (int it = 0; it < 2; ++it)
      Pa[it] = loadA16(&sP[it][l16 * 32 + half * 8]);
    __syncthreads();

    // ---- O += P @ V^T  (B operand: col = d, K = n contiguous) -----------
#pragma unroll
    for (int dc = 0; dc < 4; ++dc) {
      const v16h Bv = loadB16(vbase + (size_t)(dc * 16 + l16) * Nn + j0 + half * 8);
#pragma unroll
      for (int it = 0; it < 2; ++it)
        Oacc[it][dc] = __builtin_amdgcn_wmma_f32_16x16x32_f16(
            false, Pa[it], false, Bv, (short)0, Oacc[it][dc], false, false);
    }
  }

  // ---- finalize: out[b][h*64+dd][m] = Oacc / l ---------------------------
#pragma unroll
  for (int it = 0; it < 2; ++it) {
#pragma unroll
    for (int dc = 0; dc < 4; ++dc) {
      const int dd = dc * 16 + l16;
#pragma unroll
      for (int v = 0; v < 8; ++v) {
        const int mrow = i0 + it * 16 + v + half * 8;
        out[((size_t)b * Cc + h * Dd + dd) * Nn + mrow] = Oacc[it][dc][v] / l[it][v];
      }
    }
  }
}

// ---------------------------------------------------------------------------
// Launch
// ---------------------------------------------------------------------------
extern "C" void kernel_launch(void* const* d_in, const int* in_sizes, int n_in,
                              void* d_out, int out_size, void* d_ws, size_t ws_size,
                              hipStream_t stream) {
  const float* x     = (const float*)d_in[0];
  const float* Wq    = (const float*)d_in[1];
  const float* bq    = (const float*)d_in[2];
  const float* Wk    = (const float*)d_in[3];
  const float* bk    = (const float*)d_in[4];
  const float* Wv    = (const float*)d_in[5];
  const float* bv    = (const float*)d_in[6];
  const float* rel_h = (const float*)d_in[7];
  const float* rel_w = (const float*)d_in[8];
  float* out = (float*)d_out;

  // workspace: xT | QT | KT | Vd | relT   (all f16)
  const size_t nBNC = (size_t)Bc * Nn * Cc;             // == B*H*N*D
  _Float16* xT   = (_Float16*)d_ws;
  _Float16* QT   = xT + nBNC;
  _Float16* KT   = QT + nBNC;
  _Float16* Vd   = KT + nBNC;
  _Float16* relT = Vd + nBNC;

  k_cast_xT<<<(int)(nBNC / 256), 256, 0, stream>>>(x, xT);
  k_build_rel<<<(HEADS * Nn * Dd) / 256, 256, 0, stream>>>(rel_h, rel_w, relT);

  dim3 gp(Bc * (Nn / 32), Cc / 16, 3);
  k_proj<<<gp, 32, 0, stream>>>(Wq, bq, Wk, bk, Wv, bv, xT, QT, KT, Vd);

  k_attn<<<Bc * HEADS * (Nn / 32), 32, 0, stream>>>(QT, KT, Vd, relT, out);
}
